// MultiScaleDecoder_87299505258574
// MI455X (gfx1250) — matrix operand order repaired and checked by reading, not compiled, not run on hardware
//
#include <hip/hip_runtime.h>
#include <hip/hip_bf16.h>

// ---------------------------------------------------------------------------
// MultiScaleDecoder on gfx1250 (MI455X): GEMM chain in bf16 WMMA.
//   dense : [3*16384 x 512] x [512 x 512]   -> act1 [65536 x 384] (scatter)
//   dc1   : [65536 x 384]   x [384 x 256]   -> act2 [1048576 x 64] (scatter)
//   dc2   : [1048576 x 64]  x [64 x 128]    -> act3 [1048576 x 32] (scatter)
//   dc3   : [1048576 x 32]  x [32 x 12pad16]-> out  f32 sigmoid (scatter)
// Data movement: TDM (tensor_load_to_lds) for uniform A/B tiles in dc1/dc2,
// async global->LDS b128 for the dense B tile, VALU cvt path for dense A.
// ---------------------------------------------------------------------------

typedef __attribute__((ext_vector_type(16))) __bf16 v16bf;
typedef __attribute__((ext_vector_type(8)))  __bf16 v8bf;
typedef __attribute__((ext_vector_type(8)))  float  v8f;
typedef __attribute__((ext_vector_type(4)))  unsigned int v4u;
typedef __attribute__((ext_vector_type(8)))  int v8i;
typedef __attribute__((ext_vector_type(4)))  int v4i;

union Frag16 {
    v16bf v;
    struct { v8bf lo, hi; } h;
};

__device__ __forceinline__ v8f zero8() {
    v8f z = {0.f, 0.f, 0.f, 0.f, 0.f, 0.f, 0.f, 0.f};
    return z;
}

// Low 32 bits of a generic LDS pointer = byte offset within the wave's LDS
// allocation (flat-aperture rule).
__device__ __forceinline__ unsigned lds_off(const void* p) {
    return (unsigned)(unsigned long long)p;
}

// Async global->LDS copy, 16 bytes per lane. Tracked by ASYNCcnt.
__device__ __forceinline__ void async_copy_b128(unsigned lds, const void* g) {
    asm volatile("global_load_async_to_lds_b128 %0, %1, off"
                 :: "v"(lds), "v"(g) : "memory");
}
__device__ __forceinline__ void wait_async0() {
    asm volatile("s_wait_asynccnt 0x0" ::: "memory");
}

// TDM: load a 2D tile (tile_d1 rows x tile_d0 elems, 2-byte elements) from a
// row-major tensor with row stride stride0 (elements) into contiguous LDS.
// D# layout per CDNA5 ISA ch.8: group0 {count,lds_addr,global_addr,type=2},
// group1 {data_size, tensor dims, tile dims, dim0 stride}. This toolchain's
// builtin takes 6 args (g0, g1, g2, g3, extra group, cpol).
__device__ __forceinline__ void tdm_load_2d(
    unsigned ldsOff, const void* gptr,
    unsigned tensor_d0, unsigned tensor_d1,
    unsigned tile_d0, unsigned tile_d1,
    unsigned long long stride0) {
    const unsigned long long ga = (unsigned long long)gptr;
    v4u g0;
    g0[0] = 1u;                                             // count=1, user mode
    g0[1] = ldsOff;                                         // lds_addr
    g0[2] = (unsigned)(ga & 0xFFFFFFFFu);                   // global_addr[31:0]
    g0[3] = (unsigned)((ga >> 32) & 0x01FFFFFFu) | (2u << 30); // [56:32] | type=2
    v8i g1;
    g1[0] = (int)(1u << 16);                                // wg_mask=0, data_size=1 (2B)
    g1[1] = (int)((tensor_d0 & 0xFFFFu) << 16);             // tensor_dim0[15:0] @bits63:48
    g1[2] = (int)((tensor_d0 >> 16) | ((tensor_d1 & 0xFFFFu) << 16));
    g1[3] = (int)((tensor_d1 >> 16) | (tile_d0 << 16));     // tile_dim0 @bits127:112
    g1[4] = (int)(tile_d1 & 0xFFFFu);                       // tile_dim1, tile_dim2=0
    g1[5] = (int)(unsigned)(stride0 & 0xFFFFFFFFu);         // tensor_dim0_stride[31:0]
    g1[6] = (int)(unsigned)((stride0 >> 32) & 0xFFFFu);     // stride[47:32], dim1_stride=0
    g1[7] = 0;
    v4i g2 = {0, 0, 0, 0};
    v4i g3 = {0, 0, 0, 0};
    v8i g4 = {0, 0, 0, 0, 0, 0, 0, 0};
    __builtin_amdgcn_tensor_load_to_lds(g0, g1, g2, g3, g4, 0);
}

// A fragment (16x32 bf16, M x K). Per ISA: lane L holds row M = L&15;
// lanes 0-15 hold K = {0..7, 16..23}, lanes 16-31 hold K = {8..15, 24..31}.
__device__ __forceinline__ v16bf ldsA_frag(const __bf16* tile, int ldk, int mt, int lane) {
    const int row = mt + (lane & 15);
    const int lo  = (lane >> 4) << 3;          // 0 or 8
    Frag16 f;
    f.h.lo = *(const v8bf*)(tile + row * ldk + lo);
    f.h.hi = *(const v8bf*)(tile + row * ldk + 16 + lo);
    return f.v;
}

// B fragment (32x16 bf16, K x N), tile stored transposed in LDS as [N][K].
// Lane L holds column N = L&15; lanes 0-15 K={0..7,16..23}, lanes 16-31 K={8..15,24..31}.
__device__ __forceinline__ v16bf ldsB_frag(const __bf16* tile, int ldk, int nt, int lane) {
    const int col  = nt + (lane & 15);
    const int base = (lane >> 4) << 3;         // 0 or 8
    Frag16 f;
    f.h.lo = *(const v8bf*)(tile + col * ldk + base);
    f.h.hi = *(const v8bf*)(tile + col * ldk + base + 16);
    return f.v;
}

// ---------------------------------------------------------------------------
// Weight prep: f32 -> bf16, [N][K] layout. dense_w must transpose; the Keras
// ConvT kernels (kh,kw,f,k) flatten directly to [N][K]. w3 padded to N=16.
// ---------------------------------------------------------------------------
__global__ __launch_bounds__(256) void k_prep(
    const float* __restrict__ dw,  const float* __restrict__ w1,
    const float* __restrict__ w2,  const float* __restrict__ w3,
    __bf16* __restrict__ Wdt, __bf16* __restrict__ W1t,
    __bf16* __restrict__ W2t, __bf16* __restrict__ W3t) {
    const int idx = blockIdx.x * blockDim.x + threadIdx.x;
    if (idx < 512 * 512) {                       // dense: [k][n] -> [n][k]
        const int n = idx >> 9, k = idx & 511;
        Wdt[idx] = (__bf16)dw[k * 512 + n];
        return;
    }
    const int i1 = idx - 512 * 512;
    if (i1 < 256 * 384) { W1t[i1] = (__bf16)w1[i1]; return; }
    const int i2 = i1 - 256 * 384;
    if (i2 < 128 * 64) { W2t[i2] = (__bf16)w2[i2]; return; }
    const int i3 = i2 - 128 * 64;
    if (i3 < 16 * 32) {
        const int n = i3 >> 5;
        W3t[i3] = (n < 12) ? (__bf16)w3[i3] : (__bf16)0.0f;
    }
}

// ---------------------------------------------------------------------------
// Dense stage: A = input_s (f32 -> bf16 while staging to LDS), B via async
// global->LDS b128 copies. grid = (512/64, 16384/128, 3), block = 256.
// ---------------------------------------------------------------------------
__global__ __launch_bounds__(256) void k_dense(
    const float* __restrict__ in0, const float* __restrict__ in1,
    const float* __restrict__ in2, const __bf16* __restrict__ Wdt,
    const float* __restrict__ bias, __bf16* __restrict__ act1) {
    __shared__ __bf16 sA[128 * 32];
    __shared__ __bf16 sB[64 * 32];
    const int s = blockIdx.z;
    const float* A = (s == 0) ? in0 : (s == 1 ? in1 : in2);
    const int mBase = blockIdx.y * 128;
    const int nBase = blockIdx.x * 64;
    const int t = threadIdx.x, lane = t & 31, wave = t >> 5;
    const int wr = wave >> 1, wc = wave & 1;

    // Per-thread fixed staging slots
    const int rA = t >> 1, kA = (t & 1) << 4;      // A: 16 elems
    const int rB = t >> 2, kB = (t & 3) << 3;      // B: 8 elems (16B)
    const unsigned sBslot = lds_off(sB + rB * 32 + kB);

    v8f acc[2][2];
    acc[0][0] = zero8(); acc[0][1] = zero8();
    acc[1][0] = zero8(); acc[1][1] = zero8();

    for (int kb = 0; kb < 512; kb += 32) {
        // B tile: async DMA straight into LDS (ASYNCcnt)
        async_copy_b128(sBslot, Wdt + (size_t)(nBase + rB) * 512 + kb + kB);
        {   // A tile: f32 -> bf16 conversion through VGPRs
            const float* src = A + (size_t)(mBase + rA) * 512 + kb + kA;
            __builtin_prefetch(src + 32, 0, 1);
            __bf16* dst = sA + rA * 32 + kA;
#pragma unroll
            for (int e = 0; e < 16; ++e) dst[e] = (__bf16)src[e];
        }
        wait_async0();
        __syncthreads();
        v16bf af[2], bfg[2];
        af[0]  = ldsA_frag(sA, 32, wr * 32,      lane);
        af[1]  = ldsA_frag(sA, 32, wr * 32 + 16, lane);
        bfg[0] = ldsB_frag(sB, 32, wc * 32,      lane);
        bfg[1] = ldsB_frag(sB, 32, wc * 32 + 16, lane);
#pragma unroll
        for (int im = 0; im < 2; ++im)
#pragma unroll
            for (int jn = 0; jn < 2; ++jn)
                acc[im][jn] = __builtin_amdgcn_wmma_f32_16x16x32_bf16(
                    false, af[im], false, bfg[jn], (short)0, acc[im][jn], false, false);
        __syncthreads();
    }

    // Epilogue: relu + scatter into act1[(p*4 + i*2 + j)*384 + s*128 + k0]
#pragma unroll
    for (int im = 0; im < 2; ++im)
#pragma unroll
        for (int jn = 0; jn < 2; ++jn)
#pragma unroll
            for (int g = 0; g < 8; ++g) {
                const int m = wr * 32 + im * 16 + ((lane >> 4) << 3) + g;
                const int n = wc * 32 + jn * 16 + (lane & 15);
                const int p = mBase + m;
                const int q = nBase + n;
                float v = acc[im][jn][g] + bias[q];
                v = v > 0.f ? v : 0.f;
                const int i = q >> 8, j = (q >> 7) & 1, k0 = q & 127;
                act1[(size_t)(p * 4 + i * 2 + j) * 384 + s * 128 + k0] = (__bf16)v;
            }
}

// ---------------------------------------------------------------------------
// deconv1: [65536 x 384] x [384 x 256] -> act2 [1048576 x 64].
// A and B tiles via TDM (TENSORcnt). grid = (256/64, 65536/128), block = 256.
// ---------------------------------------------------------------------------
__global__ __launch_bounds__(256) void k_deconv1(
    const __bf16* __restrict__ act1, const __bf16* __restrict__ W1t,
    const float* __restrict__ b1, __bf16* __restrict__ act2) {
    __shared__ __bf16 sA[128 * 32];
    __shared__ __bf16 sB[64 * 32];
    const int mBase = blockIdx.y * 128;
    const int nBase = blockIdx.x * 64;
    const int t = threadIdx.x, lane = t & 31, wave = t >> 5;
    const int wr = wave >> 1, wc = wave & 1;
    const unsigned sAoff = lds_off(sA), sBoff = lds_off(sB);

    v8f acc[2][2];
    acc[0][0] = zero8(); acc[0][1] = zero8();
    acc[1][0] = zero8(); acc[1][1] = zero8();

    for (int kb = 0; kb < 384; kb += 32) {
        if (wave == 0) {
            tdm_load_2d(sAoff, act1 + (size_t)mBase * 384 + kb, 384u, 65536u, 32u, 128u, 384ull);
            tdm_load_2d(sBoff, W1t + (size_t)nBase * 384 + kb, 384u, 256u, 32u, 64u, 384ull);
            __builtin_amdgcn_s_wait_tensorcnt(0);
        }
        __syncthreads();
        v16bf af[2], bfg[2];
        af[0]  = ldsA_frag(sA, 32, wr * 32,      lane);
        af[1]  = ldsA_frag(sA, 32, wr * 32 + 16, lane);
        bfg[0] = ldsB_frag(sB, 32, wc * 32,      lane);
        bfg[1] = ldsB_frag(sB, 32, wc * 32 + 16, lane);
#pragma unroll
        for (int im = 0; im < 2; ++im)
#pragma unroll
            for (int jn = 0; jn < 2; ++jn)
                acc[im][jn] = __builtin_amdgcn_wmma_f32_16x16x32_bf16(
                    false, af[im], false, bfg[jn], (short)0, acc[im][jn], false, false);
        __syncthreads();
    }

    // row1 = p*4 + i*2 + j ; n1 = (a*2+c)*64 + f ; out row = p*16 + (2i+a)*4 + (2j+c)
#pragma unroll
    for (int im = 0; im < 2; ++im)
#pragma unroll
        for (int jn = 0; jn < 2; ++jn)
#pragma unroll
            for (int g = 0; g < 8; ++g) {
                const int m = wr * 32 + im * 16 + ((lane >> 4) << 3) + g;
                const int n = wc * 32 + jn * 16 + (lane & 15);
                const int row1 = mBase + m;
                const int n1   = nBase + n;
                const int p = row1 >> 2, i = (row1 >> 1) & 1, j = row1 & 1;
                const int a = (n1 >> 7) & 1, c = (n1 >> 6) & 1, f = n1 & 63;
                float v = acc[im][jn][g] + b1[f];
                v = v > 0.f ? v : 0.f;
                act2[(size_t)(p * 16 + (2 * i + a) * 4 + (2 * j + c)) * 64 + f] = (__bf16)v;
            }
}

// ---------------------------------------------------------------------------
// deconv2: [1048576 x 64] x [64 x 128] -> act3 [1048576 x 32].
// A and B tiles via TDM. grid = (128/64, 1048576/128), block = 256.
// ---------------------------------------------------------------------------
__global__ __launch_bounds__(256) void k_deconv2(
    const __bf16* __restrict__ act2, const __bf16* __restrict__ W2t,
    const float* __restrict__ b2, __bf16* __restrict__ act3) {
    __shared__ __bf16 sA[128 * 32];
    __shared__ __bf16 sB[64 * 32];
    const int mBase = blockIdx.y * 128;
    const int nBase = blockIdx.x * 64;
    const int t = threadIdx.x, lane = t & 31, wave = t >> 5;
    const int wr = wave >> 1, wc = wave & 1;
    const unsigned sAoff = lds_off(sA), sBoff = lds_off(sB);

    v8f acc[2][2];
    acc[0][0] = zero8(); acc[0][1] = zero8();
    acc[1][0] = zero8(); acc[1][1] = zero8();

    for (int kb = 0; kb < 64; kb += 32) {
        if (wave == 0) {
            tdm_load_2d(sAoff, act2 + (size_t)mBase * 64 + kb, 64u, 1048576u, 32u, 128u, 64ull);
            tdm_load_2d(sBoff, W2t + (size_t)nBase * 64 + kb, 64u, 128u, 32u, 64u, 64ull);
            __builtin_amdgcn_s_wait_tensorcnt(0);
        }
        __syncthreads();
        v16bf af[2], bfg[2];
        af[0]  = ldsA_frag(sA, 32, wr * 32,      lane);
        af[1]  = ldsA_frag(sA, 32, wr * 32 + 16, lane);
        bfg[0] = ldsB_frag(sB, 32, wc * 32,      lane);
        bfg[1] = ldsB_frag(sB, 32, wc * 32 + 16, lane);
#pragma unroll
        for (int im = 0; im < 2; ++im)
#pragma unroll
            for (int jn = 0; jn < 2; ++jn)
                acc[im][jn] = __builtin_amdgcn_wmma_f32_16x16x32_bf16(
                    false, af[im], false, bfg[jn], (short)0, acc[im][jn], false, false);
        __syncthreads();
    }

    // row2 = p*16 + h*4 + w ; n2 = (a*2+c)*32 + f ; out row = p*64 + (2h+a)*8 + (2w+c)
#pragma unroll
    for (int im = 0; im < 2; ++im)
#pragma unroll
        for (int jn = 0; jn < 2; ++jn)
#pragma unroll
            for (int g = 0; g < 8; ++g) {
                const int m = wr * 32 + im * 16 + ((lane >> 4) << 3) + g;
                const int n = wc * 32 + jn * 16 + (lane & 15);
                const int row2 = mBase + m;
                const int n2   = nBase + n;
                const int p = row2 >> 4, h = (row2 >> 2) & 3, w = row2 & 3;
                const int a = (n2 >> 6) & 1, c = (n2 >> 5) & 1, f = n2 & 31;
                float v = acc[im][jn][g] + b2[f];
                v = v > 0.f ? v : 0.f;
                act3[(size_t)(p * 64 + (2 * h + a) * 8 + (2 * w + c)) * 32 + f] = (__bf16)v;
            }
}

// ---------------------------------------------------------------------------
// deconv3 + sigmoid: [1048576 x 32] x [32 x 16(pad of 12)] -> f32 output.
// One 16x16x32 WMMA per wave, fragments loaded straight from global (K=32).
// grid = 1048576/128, block = 256 (8 waves x 16 rows).
// ---------------------------------------------------------------------------
__global__ __launch_bounds__(256) void k_deconv3(
    const __bf16* __restrict__ act3, const __bf16* __restrict__ W3t,
    const float* __restrict__ b3, float* __restrict__ out) {
    const int t = threadIdx.x, lane = t & 31, wave = t >> 5;
    const size_t rowBase = (size_t)blockIdx.x * 128 + (size_t)wave * 16;

    // A fragment: row = rowBase + (lane&15); K pattern {lo..lo+7, 16+lo..}
    const int lo = (lane >> 4) << 3;
    const __bf16* arow = act3 + (rowBase + (lane & 15)) * 32;
    Frag16 fa;
    fa.h.lo = *(const v8bf*)(arow + lo);
    fa.h.hi = *(const v8bf*)(arow + 16 + lo);

    // B fragment: col = lane&15, K pattern {base..base+7, 16+base..}
    const int base = (lane >> 4) << 3;
    const __bf16* brow = W3t + (size_t)(lane & 15) * 32;
    Frag16 fb;
    fb.h.lo = *(const v8bf*)(brow + base);
    fb.h.hi = *(const v8bf*)(brow + base + 16);

    v8f acc = zero8();
    acc = __builtin_amdgcn_wmma_f32_16x16x32_bf16(
        false, fa.v, false, fb.v, (short)0, acc, false, false);

    const int n = lane & 15;
    if (n < 12) {
        const int a = n / 6, c = (n / 3) & 1, ch = n % 3;
        const float bv = b3[ch];
#pragma unroll
        for (int g = 0; g < 8; ++g) {
            const size_t row = rowBase + ((lane >> 4) << 3) + g;
            const size_t p = row >> 6;
            const int pos = (int)(row & 63), h8 = pos >> 3, w8 = pos & 7;
            const int H = 2 * h8 + a, W = 2 * w8 + c;
            float v = acc[g] + bv;
            v = 1.f / (1.f + __expf(-v));
            out[p * 768 + (size_t)(H * 16 + W) * 3 + ch] = v;
        }
    }
}

// ---------------------------------------------------------------------------
extern "C" void kernel_launch(void* const* d_in, const int* in_sizes, int n_in,
                              void* d_out, int out_size, void* d_ws, size_t ws_size,
                              hipStream_t stream) {
    const float* in0   = (const float*)d_in[0];
    const float* in1   = (const float*)d_in[1];
    const float* in2   = (const float*)d_in[2];
    const float* dw    = (const float*)d_in[3];
    const float* dbias = (const float*)d_in[4];
    const float* w1    = (const float*)d_in[5];
    const float* b1    = (const float*)d_in[6];
    const float* w2    = (const float*)d_in[7];
    const float* b2    = (const float*)d_in[8];
    const float* w3    = (const float*)d_in[9];
    const float* b3    = (const float*)d_in[10];
    float* out = (float*)d_out;

    char* ws = (char*)d_ws;
    size_t off = 0;
    auto carve = [&](size_t bytes) -> void* {
        void* p = ws + off;
        off = (off + bytes + 255) & ~(size_t)255;
        return p;
    };
    __bf16* Wdt  = (__bf16*)carve(512 * 512 * sizeof(unsigned short));
    __bf16* W1t  = (__bf16*)carve(256 * 384 * sizeof(unsigned short));
    __bf16* W2t  = (__bf16*)carve(128 * 64 * sizeof(unsigned short));
    __bf16* W3t  = (__bf16*)carve(16 * 32 * sizeof(unsigned short));
    __bf16* act1 = (__bf16*)carve((size_t)65536 * 384 * sizeof(unsigned short));
    __bf16* act2 = (__bf16*)carve((size_t)1048576 * 64 * sizeof(unsigned short));
    __bf16* act3 = (__bf16*)carve((size_t)1048576 * 32 * sizeof(unsigned short));

    const int prepElems = 512 * 512 + 256 * 384 + 128 * 64 + 16 * 32;
    k_prep<<<(prepElems + 255) / 256, 256, 0, stream>>>(dw, w1, w2, w3, Wdt, W1t, W2t, W3t);

    k_dense  <<<dim3(8, 128, 3),   256, 0, stream>>>(in0, in1, in2, Wdt, dbias, act1);
    k_deconv1<<<dim3(4, 512),      256, 0, stream>>>(act1, W1t, b1, act2);
    k_deconv2<<<dim3(2, 8192),     256, 0, stream>>>(act2, W2t, b2, act3);
    k_deconv3<<<dim3(8192),        256, 0, stream>>>(act3, W3t, b3, out);
}
